// DecoderRNN_56796647523052
// MI455X (gfx1250) — matrix operand order repaired and checked by reading, not compiled, not run on hardware
//
#include <hip/hip_runtime.h>

typedef __attribute__((ext_vector_type(16))) __bf16 v16bf;
typedef __attribute__((ext_vector_type(8)))  __bf16 v8bf;
typedef __attribute__((ext_vector_type(8)))  float  v8f;

#define BB   32
#define TT   64
#define SS   128
#define HH   1024
#define EE   512
#define VV   32000
#define OOVN 20
#define VEXT (VV + OOVN)
#define G3   (3 * HH)     /* 3072 */
#define KIH  (EE + HH)    /* 1536 */
#define K2H  (2 * HH)     /* 2048 */

static __device__ __forceinline__ float sigmoidf_(float x) {
  return 1.0f / (1.0f + __expf(-x));
}

// ---------------------------------------------------------------------------
// Phase 0: f32 [N][K] -> bf16 [K][N] tiled transpose (weights become K-major
// so WMMA B loads are per-k scalar reads, adjacent n across lanes).
// ---------------------------------------------------------------------------
__global__ void transpose_to_bf16(const float* __restrict__ src,
                                  __bf16* __restrict__ dst, int Nr, int Kc) {
  __shared__ float tile[32][33];
  const int n0 = blockIdx.x * 32, k0 = blockIdx.y * 32;
  const int tx = threadIdx.x, ty = threadIdx.y;  // 32 x 8
  for (int r = ty; r < 32; r += 8) {
    int n = n0 + r, k = k0 + tx;
    tile[r][tx] = (n < Nr && k < Kc) ? src[(size_t)n * Kc + k] : 0.0f;
  }
  __syncthreads();
  for (int r = ty; r < 32; r += 8) {
    int k = k0 + r, n = n0 + tx;
    if (k < Kc && n < Nr) dst[(size_t)k * Nr + n] = (__bf16)tile[tx][r];
  }
}

__global__ void init_state(const float* __restrict__ h0, float* __restrict__ h,
                           __bf16* __restrict__ h16, __bf16* __restrict__ xcat) {
  int i = blockIdx.x * blockDim.x + threadIdx.x;
  if (i >= BB * HH) return;
  int b = i / HH, j = i % HH;
  float v = h0[i];
  h[i] = v;
  h16[i] = (__bf16)v;
  xcat[(size_t)b * KIH + EE + j] = (__bf16)0.0f;  // last_out starts at zero
}

__global__ void emb_gather(const int* __restrict__ targets, int t,
                           const float* __restrict__ embW,
                           __bf16* __restrict__ xcat) {
  const int b = blockIdx.x, tid = threadIdx.x;  // 256 threads
  int tok = targets[b * TT + t];
  if (tok >= VV) tok = 2;  // IDX_UNK
  const float* er = embW + (size_t)tok * EE;
  for (int j = tid; j < EE; j += 256)
    xcat[(size_t)b * KIH + j] = (__bf16)er[j];
}

// ---------------------------------------------------------------------------
// Wave-level M=32 x N=32 WMMA tile: C(32x32 @ n0) = A(32xK) @ B(KxN) + bias.
// bf16 operands, f32 accumulation, 4 accumulators, A fragments reused across
// the two adjacent N-tiles (4 b128 A loads + 32 scalar B loads per 4 WMMAs).
// A fragment: lane l holds row (l%16), K chunks [off..off+8) and [off+16..+8),
//             off = (l>=16)*8.
// B fragment: lane l holds column N=n0+(l%16), same K order as A (strided by ldb).
// ---------------------------------------------------------------------------
__device__ __forceinline__ void gemm_wave_32x32(
    const __bf16* __restrict__ A, int lda,
    const __bf16* __restrict__ B, int ldb,
    const float* __restrict__ bias,
    float* __restrict__ C, int ldc,
    __bf16* __restrict__ C16, int ldc16,
    int K, int act, int n0, int lane) {
  const int lm = lane & 15;
  const int lh = lane >> 4;

  union Frag { v16bf v; v8bf h[2]; };
  v8f acc00 = {}, acc01 = {}, acc10 = {}, acc11 = {};
  const __bf16* ap0 = A + (size_t)lm * lda + lh * 8;
  const __bf16* ap1 = A + (size_t)(lm + 16) * lda + lh * 8;
  const __bf16* bp  = B + (size_t)(lh * 8) * ldb + n0 + lm;
  const size_t bstep = (size_t)32 * ldb;

  for (int k0 = 0; k0 < K; k0 += 32) {
    Frag a0, a1, b0, b1;
    a0.h[0] = *(const v8bf*)(ap0 + k0);
    a0.h[1] = *(const v8bf*)(ap0 + k0 + 16);
    a1.h[0] = *(const v8bf*)(ap1 + k0);
    a1.h[1] = *(const v8bf*)(ap1 + k0 + 16);
#pragma unroll
    for (int s = 0; s < 8; ++s) {
      b0.v[s]     = bp[(size_t)s * ldb];
      b0.v[s + 8] = bp[(size_t)(s + 16) * ldb];
      b1.v[s]     = bp[(size_t)s * ldb + 16];
      b1.v[s + 8] = bp[(size_t)(s + 16) * ldb + 16];
    }
    bp += bstep;
    acc00 = __builtin_amdgcn_wmma_f32_16x16x32_bf16(false, a0.v, false, b0.v,
                                                    (short)0, acc00, false, false);
    acc10 = __builtin_amdgcn_wmma_f32_16x16x32_bf16(false, a1.v, false, b0.v,
                                                    (short)0, acc10, false, false);
    acc01 = __builtin_amdgcn_wmma_f32_16x16x32_bf16(false, a0.v, false, b1.v,
                                                    (short)0, acc01, false, false);
    acc11 = __builtin_amdgcn_wmma_f32_16x16x32_bf16(false, a1.v, false, b1.v,
                                                    (short)0, acc11, false, false);
  }

  const int nA = n0 + lm;
  const int nB = nA + 16;
  const float bvA = bias ? bias[nA] : 0.0f;
  const float bvB = bias ? bias[nB] : 0.0f;
#pragma unroll
  for (int r = 0; r < 8; ++r) {
    const int row0 = r + lh * 8;
    float v00 = acc00[r] + bvA;  // rows 0..15,  cols n0..n0+15
    float v10 = acc10[r] + bvA;  // rows 16..31, cols n0..n0+15
    float v01 = acc01[r] + bvB;  // rows 0..15,  cols n0+16..n0+31
    float v11 = acc11[r] + bvB;  // rows 16..31, cols n0+16..n0+31
    if (act) {
      v00 = tanhf(v00); v10 = tanhf(v10);
      v01 = tanhf(v01); v11 = tanhf(v11);
    }
    if (C) {
      C[(size_t)row0 * ldc + nA]        = v00;
      C[(size_t)(row0 + 16) * ldc + nA] = v10;
      C[(size_t)row0 * ldc + nB]        = v01;
      C[(size_t)(row0 + 16) * ldc + nB] = v11;
    }
    if (C16) {
      C16[(size_t)row0 * ldc16 + nA]        = (__bf16)v00;
      C16[(size_t)(row0 + 16) * ldc16 + nA] = (__bf16)v10;
      C16[(size_t)row0 * ldc16 + nB]        = (__bf16)v01;
      C16[(size_t)(row0 + 16) * ldc16 + nB] = (__bf16)v11;
    }
  }
}

// Generic single-GEMM kernel (N must be a multiple of 32).
__global__ void wmma_gemm_m32(const __bf16* __restrict__ A, int lda,
                              const __bf16* __restrict__ B, int ldb,
                              const float* __restrict__ bias,
                              float* __restrict__ C, int ldc,
                              __bf16* __restrict__ C16, int ldc16,
                              int N, int K, int act) {
  const int lane = threadIdx.x & 31;
  const int n0 = (blockIdx.x * (blockDim.x >> 5) + (threadIdx.x >> 5)) * 32;
  if (n0 >= N) return;  // wave-uniform: EXEC stays all-ones for WMMA
  gemm_wave_32x32(A, lda, B, ldb, bias, C, ldc, C16, ldc16, K, act, n0, lane);
}

// Fused GRU gate GEMMs: blockIdx.y == 0 -> gi = xcat @ WihT + bih (K=1536)
//                       blockIdx.y == 1 -> gh = h16  @ WhhT + bhh (K=1024)
__global__ void gru_gates_gemm(const __bf16* __restrict__ xcat,
                               const __bf16* __restrict__ WihT,
                               const float* __restrict__ bih,
                               float* __restrict__ gi,
                               const __bf16* __restrict__ h16,
                               const __bf16* __restrict__ WhhT,
                               const float* __restrict__ bhh,
                               float* __restrict__ gh) {
  const int lane = threadIdx.x & 31;
  const int n0 = (blockIdx.x * (blockDim.x >> 5) + (threadIdx.x >> 5)) * 32;
  if (n0 >= G3) return;
  if (blockIdx.y == 0) {
    gemm_wave_32x32(xcat, KIH, WihT, G3, bih, gi, G3, (__bf16*)nullptr, 0,
                    KIH, 0, n0, lane);
  } else {
    gemm_wave_32x32(h16, HH, WhhT, G3, bhh, gh, G3, (__bf16*)nullptr, 0,
                    HH, 0, n0, lane);
  }
}

// ---------------------------------------------------------------------------
// Per-step middle: GRU gate fusion -> h_new, attention softmax, context,
// p_gen, and builds cc = [h_new, context] (bf16) for the concat GEMM.
// One block (256 threads) per batch row.
// ---------------------------------------------------------------------------
__global__ void step_mid(const float* __restrict__ gi, const float* __restrict__ gh,
                         float* __restrict__ h, __bf16* __restrict__ h16,
                         const float* __restrict__ enc, const int* __restrict__ lens,
                         const __bf16* __restrict__ xcat,
                         const float* __restrict__ ptrW, const float* __restrict__ ptrb,
                         __bf16* __restrict__ cc, float* __restrict__ attn,
                         float* __restrict__ pgen, float* __restrict__ pgen_out) {
  const int b = blockIdx.x;
  const int tid = threadIdx.x;  // 256
  __shared__ float h_lds[HH];
  __shared__ float ctx[HH];
  __shared__ float sc[SS];
  __shared__ float red[256];

  const float* gib = gi + (size_t)b * G3;
  const float* ghb = gh + (size_t)b * G3;
  float* hb = h + (size_t)b * HH;

  for (int j = tid; j < HH; j += 256) {
    float r  = sigmoidf_(gib[j] + ghb[j]);
    float z  = sigmoidf_(gib[HH + j] + ghb[HH + j]);
    float nn = tanhf(gib[2 * HH + j] + r * ghb[2 * HH + j]);
    float hn = (1.0f - z) * nn + z * hb[j];
    hb[j] = hn;
    h16[(size_t)b * HH + j] = (__bf16)hn;
    h_lds[j] = hn;
    cc[(size_t)b * K2H + j] = (__bf16)hn;
  }
  __syncthreads();

  const int len = lens[b];
  if (tid < SS) {
    const float* e = enc + ((size_t)b * SS + tid) * HH;
    float d = 0.0f;
    for (int k = 0; k < HH; ++k) d += h_lds[k] * e[k];
    sc[tid] = (tid < len) ? d : -1e9f;
  }
  __syncthreads();
  if (tid < 64) red[tid] = fmaxf(sc[tid], sc[tid + 64]);
  __syncthreads();
  for (int off = 32; off > 0; off >>= 1) {
    if (tid < off) red[tid] = fmaxf(red[tid], red[tid + off]);
    __syncthreads();
  }
  const float mx = red[0];
  __syncthreads();
  if (tid < SS) sc[tid] = __expf(sc[tid] - mx);
  __syncthreads();
  if (tid < 64) red[tid] = sc[tid] + sc[tid + 64];
  __syncthreads();
  for (int off = 32; off > 0; off >>= 1) {
    if (tid < off) red[tid] += red[tid + off];
    __syncthreads();
  }
  const float inv = 1.0f / red[0];
  __syncthreads();
  if (tid < SS) {
    sc[tid] *= inv;
    attn[(size_t)b * SS + tid] = sc[tid];
  }
  __syncthreads();

  for (int j = tid; j < HH; j += 256) {
    float c = 0.0f;
    for (int s = 0; s < SS; ++s) c += sc[s] * enc[((size_t)b * SS + s) * HH + j];
    ctx[j] = c;
    cc[(size_t)b * K2H + HH + j] = (__bf16)c;
  }
  __syncthreads();

  float part = 0.0f;
  for (int j = tid; j < HH; j += 256)
    part += ctx[j] * ptrW[j] + h_lds[j] * ptrW[HH + j];
  for (int j = tid; j < EE; j += 256)
    part += (float)xcat[(size_t)b * KIH + j] * ptrW[2 * HH + j];
  red[tid] = part;
  __syncthreads();
  for (int off = 128; off > 0; off >>= 1) {
    if (tid < off) red[tid] += red[tid + off];
    __syncthreads();
  }
  if (tid == 0) {
    float pg = sigmoidf_(red[0] + ptrb[0]);
    pgen[b] = pg;
    pgen_out[b] = pg;
  }
}

// ---------------------------------------------------------------------------
// Vocab softmax (online max+sum) + pointer-copy scatter. Block per batch row.
// Scatter is serialized on one thread for determinism.
// ---------------------------------------------------------------------------
__global__ void out_softmax(const float* __restrict__ logits,
                            const float* __restrict__ pgen,
                            const float* __restrict__ attn,
                            const int* __restrict__ inputs,
                            float* __restrict__ out) {
  const int b = blockIdx.x;
  const int tid = threadIdx.x;  // 256
  __shared__ float rm[256], rs[256];
  const float* lb = logits + (size_t)b * VV;
  float m = -3.4e38f, s = 0.0f;
  for (int v = tid; v < VV; v += 256) {
    float x = lb[v];
    float nm = fmaxf(m, x);
    s = s * __expf(m - nm) + __expf(x - nm);
    m = nm;
  }
  rm[tid] = m; rs[tid] = s;
  __syncthreads();
  for (int off = 128; off > 0; off >>= 1) {
    if (tid < off) {
      float m2 = rm[tid + off], s2 = rs[tid + off];
      float nm = fmaxf(rm[tid], m2);
      rs[tid] = rs[tid] * __expf(rm[tid] - nm) + s2 * __expf(m2 - nm);
      rm[tid] = nm;
    }
    __syncthreads();
  }
  const float M = rm[0];
  const float inv = 1.0f / rs[0];
  const float pg = pgen[b];
  float* ob = out + (size_t)b * VEXT;
  for (int v = tid; v < VV; v += 256) ob[v] = pg * __expf(lb[v] - M) * inv;
  for (int v = VV + tid; v < VEXT; v += 256) ob[v] = 0.0f;
  __syncthreads();
  if (tid == 0) {
    const float cpg = 1.0f - pg;
    for (int sI = 0; sI < SS; ++sI) {
      int idx = inputs[b * SS + sI];
      ob[idx] += cpg * attn[(size_t)b * SS + sI];
    }
  }
}

// ---------------------------------------------------------------------------
extern "C" void kernel_launch(void* const* d_in, const int* in_sizes, int n_in,
                              void* d_out, int out_size, void* d_ws, size_t ws_size,
                              hipStream_t stream) {
  (void)in_sizes; (void)n_in; (void)out_size; (void)ws_size;
  const int*   targets = (const int*)  d_in[0];
  const float* h0      = (const float*)d_in[1];
  const float* enc     = (const float*)d_in[2];
  const int*   inputs  = (const int*)  d_in[3];
  const int*   lens    = (const int*)  d_in[4];
  const float* embW    = (const float*)d_in[6];
  const float* Wih     = (const float*)d_in[7];
  const float* Whh     = (const float*)d_in[8];
  const float* bih     = (const float*)d_in[9];
  const float* bhh     = (const float*)d_in[10];
  const float* Wc      = (const float*)d_in[11];
  const float* bc      = (const float*)d_in[12];
  const float* Wo      = (const float*)d_in[13];
  const float* bo      = (const float*)d_in[14];
  const float* ptrW    = (const float*)d_in[15];
  const float* ptrb    = (const float*)d_in[16];

  char* w = (char*)d_ws;
  auto carve = [&](size_t bytes) -> char* {
    char* p = w;
    w += (bytes + 255) & ~(size_t)255;
    return p;
  };
  __bf16* WihT   = (__bf16*)carve((size_t)KIH * G3 * 2);
  __bf16* WhhT   = (__bf16*)carve((size_t)HH * G3 * 2);
  __bf16* WcT    = (__bf16*)carve((size_t)K2H * HH * 2);
  __bf16* WoT    = (__bf16*)carve((size_t)HH * VV * 2);
  __bf16* xcat   = (__bf16*)carve((size_t)BB * KIH * 2);  // [emb | last_out] bf16
  __bf16* h16    = (__bf16*)carve((size_t)BB * HH * 2);
  float*  hbuf   = (float*) carve((size_t)BB * HH * 4);
  __bf16* cc     = (__bf16*)carve((size_t)BB * K2H * 2);  // [h_new | context]
  float*  gi     = (float*) carve((size_t)BB * G3 * 4);
  float*  gh     = (float*) carve((size_t)BB * G3 * 4);
  float*  attn   = (float*) carve((size_t)BB * SS * 4);
  float*  pgen   = (float*) carve((size_t)BB * 4);
  float*  logits = (float*) carve((size_t)BB * VV * 4);

  float* outs  = (float*)d_out;
  float* pgens = outs + (size_t)(TT - 1) * BB * VEXT;

  const dim3 tb(32, 8);
  transpose_to_bf16<<<dim3(G3 / 32, KIH / 32), tb, 0, stream>>>(Wih, WihT, G3, KIH);
  transpose_to_bf16<<<dim3(G3 / 32, HH / 32),  tb, 0, stream>>>(Whh, WhhT, G3, HH);
  transpose_to_bf16<<<dim3(HH / 32, K2H / 32), tb, 0, stream>>>(Wc,  WcT,  HH, K2H);
  transpose_to_bf16<<<dim3(VV / 32, HH / 32),  tb, 0, stream>>>(Wo,  WoT,  VV, HH);
  init_state<<<(BB * HH + 255) / 256, 256, 0, stream>>>(h0, hbuf, h16, xcat);

  for (int t = 0; t < TT - 1; ++t) {
    emb_gather<<<BB, 256, 0, stream>>>(targets, t, embW, xcat);
    // gi = [emb|last_out] @ Wih^T + bih (y=0) ; gh = h @ Whh^T + bhh (y=1)
    gru_gates_gemm<<<dim3(G3 / 128, 2), 128, 0, stream>>>(
        xcat, WihT, bih, gi, h16, WhhT, bhh, gh);
    step_mid<<<BB, 256, 0, stream>>>(gi, gh, hbuf, h16, enc, lens, xcat,
                                     ptrW, ptrb, cc, attn, pgen,
                                     pgens + (size_t)t * BB);
    // concat_out = tanh(cc @ Wc^T + bc) -> bf16 into xcat[:,512:] (next
    // last_out and A-matrix of the vocab GEMM)
    wmma_gemm_m32<<<HH / 128, 128, 0, stream>>>(cc, K2H, WcT, HH, bc,
                                                (float*)nullptr, 0, xcat + EE, KIH,
                                                HH, K2H, 1);
    // logits = concat_out @ Wo^T + bo (32x32000, K=1024) — dominant GEMM
    wmma_gemm_m32<<<VV / 128, 128, 0, stream>>>(xcat + EE, KIH, WoT, VV, bo,
                                                logits, VV, (__bf16*)nullptr, 0,
                                                VV, HH, 0);
    out_softmax<<<BB, 256, 0, stream>>>(logits, pgen, attn, inputs,
                                        outs + (size_t)t * BB * VEXT);
  }
}
